// BQuantConv1d_simple_10273561772175
// MI455X (gfx1250) — matrix operand order repaired and checked by reading, not compiled, not run on hardware
//
#include <hip/hip_runtime.h>

#define KBITS 8
#define MDIM  1024   // in_features (contraction dim)
#define NDIM  1024   // out_features
#define TDIM  8192   // B_*S_ rows

#define BM 128
#define BN 128
#define BKS 64       // K elems staged to LDS per stage (2 WMMA K-steps)
#define NSTAGE (MDIM / BKS)

typedef __attribute__((ext_vector_type(16))) __bf16 v16bf;
typedef __attribute__((ext_vector_type(8)))  float  v8f;
typedef __attribute__((ext_vector_type(4)))  int    v4i;

#if defined(__has_builtin)
#if __has_builtin(__builtin_amdgcn_global_load_async_to_lds_b128) && \
    __has_builtin(__builtin_amdgcn_s_wait_asynccnt)
#define USE_ASYNC 1
#endif
#endif
#ifndef USE_ASYNC
#define USE_ASYNC 0
#endif

__device__ __forceinline__ unsigned short f2bf(float f) {
    union { float f; unsigned int u; } x; x.f = f;
    unsigned int u = x.u;
    u += 0x7FFFu + ((u >> 16) & 1u);   // round-to-nearest-even fp32 -> bf16
    return (unsigned short)(u >> 16);
}

// 16-byte global -> LDS copy; async DMA path on CDNA5, sync fallback otherwise.
__device__ __forceinline__ void copy16_g2l(const unsigned short* __restrict__ g,
                                           unsigned short* l) {
#if USE_ASYNC
    __builtin_amdgcn_global_load_async_to_lds_b128(
        (__attribute__((address_space(1))) v4i*)(unsigned short*)g,
        (__attribute__((address_space(3))) v4i*)l,
        /*offset=*/0, /*cpol=*/0);
#else
    *(uint4*)l = *(const uint4*)g;
#endif
}

// --- Kernel 1: fold per-bit scales into one bf16 weight matrix, transposed.
// Wt[n][m] = bf16( sum_k scale[k,n] * binary[k,m,n] )
__global__ __launch_bounds__(256) void prep_w(const float* __restrict__ binary,
                                              const float* __restrict__ scale,
                                              unsigned short* __restrict__ wt) {
    const int idx = blockIdx.x * 256 + threadIdx.x;   // 0 .. M*N-1
    const int m = idx / NDIM;
    const int n = idx - m * NDIM;                     // fastest -> coalesced reads
    float acc = 0.0f;
#pragma unroll
    for (int k = 0; k < KBITS; ++k)
        acc += scale[k * NDIM + n] * binary[((size_t)k * MDIM + m) * NDIM + n];
    wt[(size_t)n * MDIM + m] = f2bf(acc);
}

// --- Kernel 2: convert x fp32 -> bf16 once (removes conversion VALU from the
// GEMM hot loop; halves A-side read traffic for the 8 N-blocks re-reading x).
__global__ __launch_bounds__(256) void prep_x(const float* __restrict__ x,
                                              unsigned short* __restrict__ xb) {
    const size_t base = ((size_t)blockIdx.x * 256 + threadIdx.x) * 8;
    const float4 a = *(const float4*)(x + base);
    const float4 b = *(const float4*)(x + base + 4);
    uint4 o;
    o.x = (unsigned int)f2bf(a.x) | ((unsigned int)f2bf(a.y) << 16);
    o.y = (unsigned int)f2bf(a.z) | ((unsigned int)f2bf(a.w) << 16);
    o.z = (unsigned int)f2bf(b.x) | ((unsigned int)f2bf(b.y) << 16);
    o.w = (unsigned int)f2bf(b.z) | ((unsigned int)f2bf(b.w) << 16);
    *(uint4*)(xb + base) = o;
}

// --- Kernel 3: out = Xb @ Wt^T + bias, bf16 WMMA GEMM, double-buffered LDS
// with async global->LDS staging.
__global__ __launch_bounds__(256) void gemm_bf16(const unsigned short* __restrict__ xb,
                                                 const unsigned short* __restrict__ wt,
                                                 const float* __restrict__ bias,
                                                 float* __restrict__ out) {
    __shared__ __align__(16) unsigned short lA[2][BM * BKS]; // [buf][row][k], 2x16KB
    __shared__ __align__(16) unsigned short lB[2][BN * BKS]; // [buf][n][k],   2x16KB

    const int tid  = threadIdx.x;
    const int lane = tid & 31;
    const int wid  = tid >> 5;        // 0..7 (8 waves)
    const int wrow = wid & 3;         // 4 wave rows  -> 32 output rows each
    const int wcol = wid >> 2;        // 2 wave cols  -> 64 output cols each

    const int tBase = blockIdx.y * BM;
    const int nBase = blockIdx.x * BN;

    const int halfsel = lane >> 4;    // 0: lanes 0-15, 1: lanes 16-31
    const int l15     = lane & 15;

    v8f acc[2][4];
#pragma unroll
    for (int i = 0; i < 2; ++i)
#pragma unroll
        for (int j = 0; j < 4; ++j)
            acc[i][j] = (v8f){0.f, 0.f, 0.f, 0.f, 0.f, 0.f, 0.f, 0.f};

    union Frag { uint4 q[2]; v16bf v; };

    // Stage issuer: 128 rows x 64 bf16 = 16 KB per tile; 256 threads x 4
    // 16B-chunks each for A and for B -> 8 async transfers per thread per stage.
    auto issue_stage = [&](int k0, int buf) {
#pragma unroll
        for (int t = 0; t < 4; ++t) {
            const int c    = tid + t * 256;   // chunk 0..1023 (8 chunks per 128B row)
            const int row  = c >> 3;
            const int col8 = (c & 7) * 8;
            copy16_g2l(xb + (size_t)(tBase + row) * MDIM + k0 + col8,
                       &lA[buf][row * BKS + col8]);
        }
#pragma unroll
        for (int t = 0; t < 4; ++t) {
            const int c    = tid + t * 256;
            const int n    = c >> 3;
            const int col8 = (c & 7) * 8;
            copy16_g2l(wt + (size_t)(nBase + n) * MDIM + k0 + col8,
                       &lB[buf][n * BKS + col8]);
        }
    };

    issue_stage(0, 0);

    for (int s = 0; s < NSTAGE; ++s) {
        const int buf = s & 1;
        if (s + 1 < NSTAGE) {
            issue_stage((s + 1) * BKS, buf ^ 1);   // overlap DMA with compute
#if USE_ASYNC
            __builtin_amdgcn_s_wait_asynccnt(8);   // retire stage s (in-order)
#endif
        } else {
#if USE_ASYNC
            __builtin_amdgcn_s_wait_asynccnt(0);
#endif
        }
        __syncthreads();   // all waves' stage-s data visible in LDS

        // 2 WMMA K-steps per stage, 8 WMMAs each
#pragma unroll
        for (int kk = 0; kk < BKS; kk += 32) {
            // A fragments (ISA 16-bit A 16x32: half-wave K split 0..7/16..23 vs 8..15/24..31)
            Frag af[2];
            const int ako = kk + (halfsel ? 8 : 0);
#pragma unroll
            for (int i = 0; i < 2; ++i) {
                const unsigned short* ap = &lA[buf][(wrow * 32 + i * 16 + l15) * BKS];
                af[i].q[0] = *(const uint4*)(ap + ako);
                af[i].q[1] = *(const uint4*)(ap + ako + 16);
            }
            // B fragments (lane = N, half-wave K split 0..7/16..23 vs 8..15/24..31)
            Frag bfr[4];
            const int bko = kk + (halfsel ? 8 : 0);
#pragma unroll
            for (int j = 0; j < 4; ++j) {
                const unsigned short* bp = &lB[buf][(wcol * 64 + j * 16 + l15) * BKS];
                bfr[j].q[0] = *(const uint4*)(bp + bko);
                bfr[j].q[1] = *(const uint4*)(bp + bko + 16);
            }
#pragma unroll
            for (int i = 0; i < 2; ++i)
#pragma unroll
                for (int j = 0; j < 4; ++j)
                    acc[i][j] = __builtin_amdgcn_wmma_f32_16x16x32_bf16(
                        false, af[i].v, false, bfr[j].v,
                        (short)0, acc[i][j], false, false);
        }

        __syncthreads();   // WAR: next iter's issue overwrites buf
    }

    // Epilogue: C layout VGPR r -> M=r (lanes 0-15) / M=r+8 (lanes 16-31); bias add.
#pragma unroll
    for (int j = 0; j < 4; ++j) {
        const int ncol = nBase + wcol * 64 + j * 16 + l15;
        const float bv = bias[ncol];
#pragma unroll
        for (int i = 0; i < 2; ++i) {
#pragma unroll
            for (int r = 0; r < 8; ++r) {
                const int trow = tBase + wrow * 32 + i * 16 + halfsel * 8 + r;
                out[(size_t)trow * NDIM + ncol] = acc[i][j][r] + bv;
            }
        }
    }
}

extern "C" void kernel_launch(void* const* d_in, const int* in_sizes, int n_in,
                              void* d_out, int out_size, void* d_ws, size_t ws_size,
                              hipStream_t stream) {
    const float* x      = (const float*)d_in[0];  // [4,2048,1024]
    const float* binary = (const float*)d_in[1];  // [8,1024,1024]
    const float* scale  = (const float*)d_in[2];  // [8,1,1024]
    const float* bias   = (const float*)d_in[3];  // [1024]
    float* out = (float*)d_out;                   // [4,2048,1024]

    unsigned short* wt = (unsigned short*)d_ws;                         // 2 MB
    unsigned short* xbuf = (unsigned short*)d_ws + (size_t)NDIM * MDIM; // 16 MB

    prep_w<<<(MDIM * NDIM) / 256, 256, 0, stream>>>(binary, scale, wt);
    prep_x<<<((size_t)TDIM * MDIM) / (256 * 8), 256, 0, stream>>>(x, xbuf);

    dim3 grid(NDIM / BN, TDIM / BM);
    gemm_bf16<<<grid, 256, 0, stream>>>(xbuf, wt, bias, out);
}
